// GATv2_66030827209333
// MI455X (gfx1250) — matrix operand order repaired and checked by reading, not compiled, not run on hardware
//
#include <hip/hip_runtime.h>

#define N_NODES  50000
#define N_EDGES  1200000
#define IN_DIM   128
#define EDGE_DIM 64
#define EMBED    64      // H*D = 8*8
#define NHEADS   8

// per-lane fragment stride in bf16 units: 16 data + 8 pad = 48 bytes
#define FRAG_STRIDE 24

typedef __attribute__((ext_vector_type(16))) __bf16 v16bf;
typedef __attribute__((ext_vector_type(8)))  float  v8f;

__device__ __forceinline__ __bf16 f2bf(float f) { return (__bf16)f; }

// order-preserving float<->uint encoding for atomic max over signed floats
__device__ __forceinline__ unsigned fenc(float f) {
    unsigned u = __float_as_uint(f);
    return (u & 0x80000000u) ? ~u : (u | 0x80000000u);
}
__device__ __forceinline__ float fdec(unsigned u) {
    return __uint_as_float((u & 0x80000000u) ? (u & 0x7FFFFFFFu) : ~u);
}

// mish(x) = x * tanh(softplus(x)) = x * (w^2-1)/(w^2+1), w = 1+e^x
__device__ __forceinline__ float mishf(float x) {
    if (x > 20.f) return x;
    float w  = 1.f + __expf(x);
    float w2 = w * w;
    return x * (w2 - 1.f) / (w2 + 1.f);
}

// Load a 16x32 bf16 A fragment (ISA layout: lanes0-15 rows M with K 0..7/16..23;
// lanes16-31 rows M with K 8..15/24..31) from a row-major fp32 row pointer.
__device__ __forceinline__ v16bf load_afrag_f32row(const float* rowp, int kc, int lane) {
    int off = (lane >> 4) * 8;
    const float* p = rowp + kc * 32 + off;
    float4 q0 = *(const float4*)(p);
    float4 q1 = *(const float4*)(p + 4);
    float4 q2 = *(const float4*)(p + 16);
    float4 q3 = *(const float4*)(p + 20);
    v16bf a;
    a[0]  = f2bf(q0.x); a[1]  = f2bf(q0.y); a[2]  = f2bf(q0.z); a[3]  = f2bf(q0.w);
    a[4]  = f2bf(q1.x); a[5]  = f2bf(q1.y); a[6]  = f2bf(q1.z); a[7]  = f2bf(q1.w);
    a[8]  = f2bf(q2.x); a[9]  = f2bf(q2.y); a[10] = f2bf(q2.z); a[11] = f2bf(q2.w);
    a[12] = f2bf(q3.x); a[13] = f2bf(q3.y); a[14] = f2bf(q3.z); a[15] = f2bf(q3.w);
    return a;
}

// Cooperative fill: pre-swizzle fp32 weight matrix W[K][EMBED] into LDS in
// per-lane B-fragment order so loads are contiguous 32B runs (2x ds_load_b128).
// Fragment (kc,t), lane l, element e  <=  W[(kc*32 + (l>>4)*8 + (e<8 ? e : 8+e)) * EMBED + t*16 + (l&15)]
__device__ __forceinline__ void fill_bfrags(const float* __restrict__ W, __bf16* Wfrag,
                                            int nkc, int tid, int nthreads) {
    int total = nkc * 4 * 32 * 16;
    for (int idx = tid; idx < total; idx += nthreads) {
        int f = idx >> 9;           // fragment id = kc*4 + t
        int r = idx & 511;
        int l = r >> 4;             // lane
        int e = r & 15;             // element
        int kc = f >> 2, t = f & 3;
        int k = kc * 32 + ((l >> 4) << 3) + (e < 8 ? e : 8 + e);
        int n = t * 16 + (l & 15);
        Wfrag[f * (32 * FRAG_STRIDE) + l * FRAG_STRIDE + e] = f2bf(W[k * EMBED + n]);
    }
}

__device__ __forceinline__ v16bf get_bfrag(const __bf16* Wfrag, int kc, int t, int lane) {
    return *(const v16bf*)(Wfrag + ((kc * 4 + t) * 32 + lane) * FRAG_STRIDE);
}

// ---------------- init: zero output / denom, reset encoded max ----------------
__global__ void k_init(float* out, unsigned* maxenc, float* denom) {
    int i = blockIdx.x * blockDim.x + threadIdx.x;
    if (i < N_NODES * EMBED) out[i] = 0.f;
    if (i < N_NODES * NHEADS) { maxenc[i] = 0u; denom[i] = 0.f; }
}

// ---------------- node projection: nodes = X @ W + b (bf16 WMMA) ----------------
__global__ __launch_bounds__(128) void k_node_gemm(
    const float* __restrict__ X, const float* __restrict__ W,
    const float* __restrict__ bias, float* __restrict__ nodes) {
    __shared__ __bf16 Wfrag[16 * 32 * FRAG_STRIDE];   // 24 KB, fragment-order
    int tid = threadIdx.x;
    fill_bfrags(W, Wfrag, 4, tid, 128);
    __syncthreads();

    int lane  = tid & 31;
    int mtile = blockIdx.x * 4 + (tid >> 5);
    if (mtile >= N_NODES / 16) return;

    const float* rowp = X + (size_t)(mtile * 16 + (lane & 15)) * IN_DIM;

    v8f acc[4] = {};
#pragma unroll
    for (int kc = 0; kc < 4; ++kc) {
        v16bf af = load_afrag_f32row(rowp, kc, lane);
#pragma unroll
        for (int t = 0; t < 4; ++t) {
            v16bf bf = get_bfrag(Wfrag, kc, t, lane);
            acc[t] = __builtin_amdgcn_wmma_f32_16x16x32_bf16(false, af, false, bf,
                                                             (short)0, acc[t], false, false);
        }
    }

    int half = lane >> 4, col0 = lane & 15;
#pragma unroll
    for (int t = 0; t < 4; ++t) {
        float wb = bias[t * 16 + col0];
#pragma unroll
        for (int j = 0; j < 8; ++j) {
            int m = mtile * 16 + j + half * 8;
            nodes[(size_t)m * EMBED + t * 16 + col0] = acc[t][j] + wb;
        }
    }
}

// ---- edge pass 1: WMMA edge proj + gather + mish + head logits + seg-max ----
__global__ __launch_bounds__(256) void k_edge_pass1(
    const float* __restrict__ EF, const float* __restrict__ We,
    const float* __restrict__ web, const float* __restrict__ avec,
    const int* __restrict__ senders, const int* __restrict__ receivers,
    const float* __restrict__ nodes, float* __restrict__ logits,
    unsigned* __restrict__ maxenc) {
    __shared__ __bf16 Wefrag[8 * 32 * FRAG_STRIDE];   // 12 KB, fragment-order
    int tid = threadIdx.x;
    fill_bfrags(We, Wefrag, 2, tid, 256);
    __syncthreads();

    int lane = tid & 31;
    int tile = blockIdx.x * 8 + (tid >> 5);     // 16 edges per tile
    size_t ebase = (size_t)tile * 16;

    // A fragments: 16 edge rows x 64 K (bf16), two K-chunks
    const float* rowp = EF + (ebase + (lane & 15)) * EDGE_DIM;
    v16bf af0 = load_afrag_f32row(rowp, 0, lane);
    v16bf af1 = load_afrag_f32row(rowp, 1, lane);

    int half = lane >> 4, col0 = lane & 15;
    // sender/receiver ids for the 8 output rows this lane owns (C layout)
    int sidx[8], ridx[8];
#pragma unroll
    for (int j = 0; j < 8; ++j) {
        int m = j + half * 8;
        sidx[j] = senders[ebase + m];
        ridx[j] = receivers[ebase + m];
    }

    v8f acc[4] = {};
#pragma unroll
    for (int t = 0; t < 4; ++t) {
        acc[t] = __builtin_amdgcn_wmma_f32_16x16x32_bf16(false, af0, false,
                     get_bfrag(Wefrag, 0, t, lane), (short)0, acc[t], false, false);
    }
#pragma unroll
    for (int t = 0; t < 4; ++t) {
        acc[t] = __builtin_amdgcn_wmma_f32_16x16x32_bf16(false, af1, false,
                     get_bfrag(Wefrag, 1, t, lane), (short)0, acc[t], false, false);
    }

#pragma unroll
    for (int t = 0; t < 4; ++t) {
        int   col = t * 16 + col0;
        float wb  = web[col];
        float av  = avec[col];           // a flattened: col = h*8 + d
#pragma unroll
        for (int j = 0; j < 8; ++j) {
            float v = acc[t][j] + wb
                    + nodes[(size_t)sidx[j] * EMBED + col]
                    + nodes[(size_t)ridx[j] * EMBED + col];
            float contrib = mishf(v) * av;
            // reduce over head dim d = col%8 (lanes differing in bits 0..2)
            contrib += __shfl_xor(contrib, 1, 32);
            contrib += __shfl_xor(contrib, 2, 32);
            contrib += __shfl_xor(contrib, 4, 32);
            if ((lane & 7) == 0) {
                int h = 2 * t + ((lane >> 3) & 1);
                size_t e = ebase + j + ((lane >> 4) << 3);
                logits[e * NHEADS + h] = contrib;
                atomicMax(&maxenc[(size_t)ridx[j] * NHEADS + h], fenc(contrib));
            }
        }
    }
}

// ---------------- edge pass 2: e = exp(logit - max), denom += e ----------------
__global__ void k_edge_pass2(const int* __restrict__ receivers,
                             float* __restrict__ logits,
                             const unsigned* __restrict__ maxenc,
                             float* __restrict__ denom) {
    int idx = blockIdx.x * blockDim.x + threadIdx.x;
    if (idx >= N_EDGES * NHEADS) return;
    int e = idx >> 3, h = idx & 7;
    int r = receivers[e];
    float mx = fdec(maxenc[(size_t)r * NHEADS + h]);
    float w  = __expf(logits[idx] - mx);
    logits[idx] = w;
    atomicAdd(&denom[(size_t)r * NHEADS + h], w);
}

__global__ void k_inv_denom(float* denom) {
    int i = blockIdx.x * blockDim.x + threadIdx.x;
    if (i < N_NODES * NHEADS) {
        float d = denom[i];
        denom[i] = (d > 0.f) ? 1.f / d : 0.f;
    }
}

// ------------- edge pass 3: out[recv] += w * nodes[sender], 4 cols/thread -------------
__global__ void k_edge_pass3(const int* __restrict__ senders,
                             const int* __restrict__ receivers,
                             const float* __restrict__ ew,
                             const float* __restrict__ inv_denom,
                             const float* __restrict__ nodes,
                             float* __restrict__ out) {
    int idx = blockIdx.x * blockDim.x + threadIdx.x;
    if (idx >= N_EDGES * 16) return;
    int e = idx >> 4, q = idx & 15, c = q * 4, h = q >> 1;
    int r = receivers[e], s = senders[e];
    float w = ew[(size_t)e * NHEADS + h] * inv_denom[(size_t)r * NHEADS + h];
    float4 sv = *(const float4*)(nodes + (size_t)s * EMBED + c);
    float* o  = out + (size_t)r * EMBED + c;
    atomicAdd(o + 0, w * sv.x);
    atomicAdd(o + 1, w * sv.y);
    atomicAdd(o + 2, w * sv.z);
    atomicAdd(o + 3, w * sv.w);
}

extern "C" void kernel_launch(void* const* d_in, const int* in_sizes, int n_in,
                              void* d_out, int out_size, void* d_ws, size_t ws_size,
                              hipStream_t stream) {
    const float* node_features = (const float*)d_in[0];
    const float* edge_features = (const float*)d_in[1];
    // d_in[2] = global_features (unused by reference math)
    const int*   senders   = (const int*)d_in[3];
    const int*   receivers = (const int*)d_in[4];
    const float* W_kernel  = (const float*)d_in[5];
    const float* W_bias    = (const float*)d_in[6];
    const float* We_kernel = (const float*)d_in[7];
    const float* We_bias   = (const float*)d_in[8];
    const float* a         = (const float*)d_in[9];
    float* out = (float*)d_out;

    // workspace layout
    float*    nodes  = (float*)d_ws;                              // N*64
    float*    logits = nodes + (size_t)N_NODES * EMBED;           // E*8 (reused as e-weights)
    unsigned* maxenc = (unsigned*)(logits + (size_t)N_EDGES * NHEADS); // N*8
    float*    denom  = (float*)(maxenc + (size_t)N_NODES * NHEADS);    // N*8

    k_init<<<(N_NODES * EMBED + 255) / 256, 256, 0, stream>>>(out, maxenc, denom);

    int ntiles = N_NODES / 16;                 // 3125
    k_node_gemm<<<(ntiles + 3) / 4, 128, 0, stream>>>(node_features, W_kernel, W_bias, nodes);

    int etiles = N_EDGES / 16;                 // 75000, 8 waves/block
    k_edge_pass1<<<etiles / 8, 256, 0, stream>>>(edge_features, We_kernel, We_bias, a,
                                                 senders, receivers, nodes, logits, maxenc);

    k_edge_pass2<<<(N_EDGES * NHEADS + 255) / 256, 256, 0, stream>>>(receivers, logits,
                                                                     maxenc, denom);
    k_inv_denom<<<(N_NODES * NHEADS + 255) / 256, 256, 0, stream>>>(denom);

    k_edge_pass3<<<(N_EDGES * 16 + 255) / 256, 256, 0, stream>>>(senders, receivers,
                                                                 logits, denom, nodes, out);
}
